// RectifyPrompter_83588653515179
// MI455X (gfx1250) — matrix operand order repaired and checked by reading, not compiled, not run on hardware
//
#include <hip/hip_runtime.h>
#include <math.h>

// ---------------- geometry ----------------
constexpr int BATCH = 16;
constexpr int NPTS  = 8192;
constexpr int NC    = 128;
constexpr int HID   = 384;
constexpr int NG    = 32;   // NUM_GROUP
constexpr int TOPD  = 12;
constexpr float BN_EPS = 1e-5f;

// ---------------- WMMA helpers ----------------
typedef __attribute__((ext_vector_type(16))) _Float16 v16h;
typedef __attribute__((ext_vector_type(8)))  _Float16 v8h;
typedef __attribute__((ext_vector_type(8)))  float    v8f;

// packed weight offsets (halves). each 32x16 K-N block = 512 halves.
constexpr int OFF_ABS1 = 0;      // Kp=384 Np=64 : 48 blocks
constexpr int OFF_ABS2 = 24576;  // 64x32 : 4
constexpr int OFF_ABS3 = 26624;  // 32x16 : 1
constexpr int OFF_P21  = 27136;  // 32x64 : 4
constexpr int OFF_P22  = 29184;  // 64x32 : 4
constexpr int OFF_P11  = 31232;  // 64x32 : 4
constexpr int OFF_P12  = 33280;  // 32x32 : 2
constexpr int OFF_H1   = 34304;  // 32x64 : 4
constexpr int OFF_H2   = 36352;  // 64x16 : 2  (total 37376 halves)

// bias offsets (floats)
constexpr int BO_ABS1 = 0,  BO_ABS2 = 64,  BO_ABS3 = 96,  BO_P21 = 112;
constexpr int BO_P22 = 176, BO_P11 = 208,  BO_P12 = 240,  BO_H1 = 272, BO_H2 = 336;

// workspace byte offsets
constexpr size_t WS_WPACK  = 0;          // 74752 B of f16
constexpr size_t WS_BIAS   = 80u * 1024; // 352 f32
constexpr size_t WS_C2POS  = 84u * 1024; // 16*32*3 f32
constexpr size_t WS_C2FEAT = 92u * 1024; // 16*32*12 f32
constexpr size_t WS_C1F    = 120u * 1024;// 16*128*32 f32

// A operand: 16x32 f16 tile from row-major [16][ldk] LDS buffer.
// lane<16: row=lane, elems 0..7 = K(kbase+0..7), 8..15 = K(kbase+16..23)
// lane>=16: row=lane-16, elems 0..7 = K(kbase+8..15), 8..15 = K(kbase+24..31)
__device__ __forceinline__ v16h ld_A(const _Float16* tile, int ldk, int kbase, int lane) {
  int row = lane & 15;
  int khi = (lane & 16) ? 8 : 0;
  const _Float16* p = tile + row * ldk + kbase + khi;
  v16h a;
  *(v8h*)&a       = *(const v8h*)p;
  *((v8h*)&a + 1) = *(const v8h*)(p + 16);
  return a;
}

// B operand: pre-packed in lane-major order, 32B contiguous per lane.
__device__ __forceinline__ v16h ld_B(const _Float16* blk, int lane) {
  return *(const v16h*)(blk + lane * 16);
}

// C initialized with per-output-channel bias broadcast across the 8 rows.
__device__ __forceinline__ v8f mk_C(const float* bias, int nbase, int lane) {
  float bv = bias[nbase + (lane & 15)];
  v8f c;
#pragma unroll
  for (int r = 0; r < 8; ++r) c[r] = bv;
  return c;
}

__device__ __forceinline__ v8f wmma32(v16h a, v16h b, v8f c) {
  return __builtin_amdgcn_wmma_f32_16x16x32_f16(false, a, false, b, (short)0, c,
                                                false, false);
}

// D writeback with ReLU to row-major [16][ldn] f16 LDS buffer.
// lane<16: col=lane, rows 0..7 ; lane>=16: col=lane-16, rows 8..15
__device__ __forceinline__ void st_D_relu(_Float16* dst, int ldn, int nbase, int lane, v8f d) {
  int col = lane & 15;
  int rbase = (lane & 16) ? 8 : 0;
#pragma unroll
  for (int r = 0; r < 8; ++r)
    dst[(rbase + r) * ldn + nbase + col] = (_Float16)fmaxf(d[r], 0.f);
}

template <int Kp, int Np>
__device__ __forceinline__ void run_layer(const _Float16* in, _Float16* out,
                                          const _Float16* wblk, const float* bvec,
                                          int lane) {
  constexpr int NT = Np / 16, KC = Kp / 32;
#pragma unroll
  for (int nt = 0; nt < NT; ++nt) {
    v8f c = mk_C(bvec, nt * 16, lane);
#pragma unroll
    for (int kc = 0; kc < KC; ++kc)
      c = wmma32(ld_A(in, Kp, kc * 32, lane), ld_B(wblk + (kc * NT + nt) * 512, lane), c);
    st_D_relu(out, Np, nt * 16, lane, c);
  }
}

// ---------------- weight packing (BN folded, B-matrix lane layout) ----------------
__global__ void pack_kernel(const float* __restrict__ W, const float* __restrict__ b,
                            const float* __restrict__ beta, const float* __restrict__ g,
                            const float* __restrict__ m, const float* __restrict__ v,
                            int cin, int cout, int Kp, int Np, int has_bn,
                            _Float16* __restrict__ wdst, float* __restrict__ bdst) {
  int total = Kp * Np;
  int ntiles = Np >> 4;
  for (int t = blockIdx.x * blockDim.x + threadIdx.x; t < total;
       t += gridDim.x * blockDim.x) {
    int i = t & 15;
    int lane = (t >> 4) & 31;
    int blk = t >> 9;
    int kc = blk / ntiles, nt = blk - kc * ntiles;
    int n = nt * 16 + (lane & 15);
    int k = kc * 32 + (i & 7) + ((lane & 16) ? 8 : 0) + ((i & 8) ? 16 : 0);
    float val = 0.f;
    if (k < cin && n < cout) {
      float s = has_bn ? g[n] * rsqrtf(v[n] + BN_EPS) : 1.f;
      val = W[n * cin + k] * s;
    }
    wdst[t] = (_Float16)val;
  }
  int t = blockIdx.x * blockDim.x + threadIdx.x;
  if (t < Np) {
    float bv = 0.f;
    if (t < cout) {
      if (has_bn) {
        float s = g[t] * rsqrtf(v[t] + BN_EPS);
        bv = (b[t] - m[t]) * s + beta[t];
      } else {
        bv = b[t];
      }
    }
    bdst[t] = bv;
  }
}

// ---------------- farthest point sampling ----------------
__global__ void fps_kernel(const float* __restrict__ c1, float* __restrict__ c2pos) {
  __shared__ float rv[NC];
  __shared__ int ri[NC];
  int b = blockIdx.x, t = threadIdx.x;
  const float* P = c1 + (size_t)b * NC * 3;
  float x = P[t * 3], y = P[t * 3 + 1], z = P[t * 3 + 2];
  float dmin = 1e10f;
  int last = 0;
  for (int it = 0; it < NG; ++it) {
    if (t == 0) {
      c2pos[((size_t)b * NG + it) * 3 + 0] = P[last * 3 + 0];
      c2pos[((size_t)b * NG + it) * 3 + 1] = P[last * 3 + 1];
      c2pos[((size_t)b * NG + it) * 3 + 2] = P[last * 3 + 2];
    }
    float lx = P[last * 3], ly = P[last * 3 + 1], lz = P[last * 3 + 2];
    float dx = x - lx, dy = y - ly, dz = z - lz;
    dmin = fminf(dmin, dx * dx + dy * dy + dz * dz);
    rv[t] = dmin; ri[t] = t;
    __syncthreads();
    for (int s = NC / 2; s > 0; s >>= 1) {
      if (t < s) {
        if (rv[t + s] > rv[t] || (rv[t + s] == rv[t] && ri[t + s] < ri[t])) {
          rv[t] = rv[t + s]; ri[t] = ri[t + s];
        }
      }
      __syncthreads();
    }
    last = ri[0];
    __syncthreads();
  }
}

// ---------------- set abstraction: knn + MLP(384,64,32,12) + maxpool ----------------
__global__ void abs_kernel(const float* __restrict__ c1, const float* __restrict__ c1feat,
                           const float* __restrict__ c2pos,
                           const _Float16* __restrict__ wpack,
                           const float* __restrict__ biases, float* __restrict__ c2feat) {
  __shared__ float sd[NC];
  __shared__ int knn[16];
  __shared__ alignas(32) _Float16 act[16 * HID];
  __shared__ alignas(32) _Float16 a2[16 * 64];
  __shared__ alignas(32) _Float16 a3[16 * 32];
  __shared__ float sm[32];

  int bg = blockIdx.x;            // b*32+g
  int b = bg >> 5;
  int lane = threadIdx.x;

  float cx = c2pos[(size_t)bg * 3], cy = c2pos[(size_t)bg * 3 + 1],
        cz = c2pos[(size_t)bg * 3 + 2];
  for (int j = lane; j < NC; j += 32) {
    float dx = cx - c1[((size_t)b * NC + j) * 3];
    float dy = cy - c1[((size_t)b * NC + j) * 3 + 1];
    float dz = cz - c1[((size_t)b * NC + j) * 3 + 2];
    sd[j] = dx * dx + dy * dy + dz * dz;
  }
  __syncthreads();
  if (lane == 0) {
    for (int k = 0; k < 16; ++k) {
      float best = 3.4e38f; int bj = 0;
      for (int j = 0; j < NC; ++j)
        if (sd[j] < best) { best = sd[j]; bj = j; }
      knn[k] = bj; sd[bj] = 3.4e38f;
    }
  }
  __syncthreads();

  for (int k = 0; k < 16; ++k) {
    const float* src = c1feat + ((size_t)b * NC + knn[k]) * HID;
    for (int ch = lane; ch < HID; ch += 32) act[k * HID + ch] = (_Float16)src[ch];
  }
  __syncthreads();

  run_layer<HID, 64>(act, a2, wpack + OFF_ABS1, biases + BO_ABS1, lane);
  run_layer<64, 32>(a2, a3, wpack + OFF_ABS2, biases + BO_ABS2, lane);

  // final 32 -> 16(12) layer, relu + maxpool over the 16 group members
  v8f c = mk_C(biases + BO_ABS3, 0, lane);
  c = wmma32(ld_A(a3, 32, 0, lane), ld_B(wpack + OFF_ABS3, lane), c);
  float mx = 0.f;
#pragma unroll
  for (int r = 0; r < 8; ++r) mx = fmaxf(mx, c[r]);   // relu folded: values >= 0
  sm[lane] = mx;
  __syncthreads();
  if (lane < TOPD)
    c2feat[(size_t)bg * TOPD + lane] = fmaxf(sm[lane], sm[lane + 16]);
}

// ---------------- propagation2: IDW interp(12) + MLP(12,64,32) ----------------
__global__ void prop2_kernel(const float* __restrict__ c1, const float* __restrict__ c2pos,
                             const float* __restrict__ c2feat,
                             const _Float16* __restrict__ wpack,
                             const float* __restrict__ biases, float* __restrict__ c1f) {
  __shared__ float s_c2[NG * 3];
  __shared__ float s_cf[NG * TOPD];
  __shared__ alignas(32) _Float16 feat[16 * 32];
  __shared__ alignas(32) _Float16 a2[16 * 64];

  int b = blockIdx.x >> 3;
  int t0 = (blockIdx.x & 7) * 16;
  int lane = threadIdx.x;

  for (int i = lane; i < NG * 3; i += 32) s_c2[i] = c2pos[(size_t)b * NG * 3 + i];
  for (int i = lane; i < NG * TOPD; i += 32) s_cf[i] = c2feat[(size_t)b * NG * TOPD + i];
  for (int i = lane; i < 16 * 32; i += 32) feat[i] = (_Float16)0.f;
  __syncthreads();

  if (lane < 16) {
    int p = t0 + lane;
    float px = c1[((size_t)b * NC + p) * 3], py = c1[((size_t)b * NC + p) * 3 + 1],
          pz = c1[((size_t)b * NC + p) * 3 + 2];
    float bd[16]; int bi[16];
    for (int k = 0; k < 16; ++k) { bd[k] = 3.4e38f; bi[k] = 0; }
    for (int j = 0; j < NG; ++j) {
      float dx = px - s_c2[j * 3], dy = py - s_c2[j * 3 + 1], dz = pz - s_c2[j * 3 + 2];
      float d = dx * dx + dy * dy + dz * dz;
      if (d < bd[15]) {
        int pos = 15;
        while (pos > 0 && bd[pos - 1] > d) {
          bd[pos] = bd[pos - 1]; bi[pos] = bi[pos - 1]; --pos;
        }
        bd[pos] = d; bi[pos] = j;
      }
    }
    float w[16], ws = 0.f;
    for (int k = 0; k < 16; ++k) { w[k] = 1.f / (bd[k] + 1e-4f); ws += w[k]; }
    float inv = 1.f / ws;
    float acc[TOPD];
    for (int cch = 0; cch < TOPD; ++cch) acc[cch] = 0.f;
    for (int k = 0; k < 16; ++k) {
      float ww = w[k] * inv;
      const float* f = &s_cf[bi[k] * TOPD];
      for (int cch = 0; cch < TOPD; ++cch) acc[cch] += ww * f[cch];
    }
    for (int cch = 0; cch < TOPD; ++cch) feat[lane * 32 + cch] = (_Float16)acc[cch];
  }
  __syncthreads();

  run_layer<32, 64>(feat, a2, wpack + OFF_P21, biases + BO_P21, lane);

  // 64 -> 32, relu, write f32 to workspace c1f[b][p][32]
#pragma unroll
  for (int nt = 0; nt < 2; ++nt) {
    v8f c = mk_C(biases + BO_P22, nt * 16, lane);
#pragma unroll
    for (int kc = 0; kc < 2; ++kc)
      c = wmma32(ld_A(a2, 64, kc * 32, lane), ld_B(wpack + OFF_P22 + (kc * 2 + nt) * 512, lane), c);
    int col = lane & 15, rbase = (lane & 16) ? 8 : 0;
#pragma unroll
    for (int r = 0; r < 8; ++r)
      c1f[((size_t)b * NC + t0 + rbase + r) * 32 + nt * 16 + col] = fmaxf(c[r], 0.f);
  }
}

// ---------------- fused tail: PE + IDW interp(32) + prop1 + head ----------------
__global__ void tail_kernel(const float* __restrict__ x, const float* __restrict__ c1,
                            const float* __restrict__ c1f,
                            const _Float16* __restrict__ wpack,
                            const float* __restrict__ biases, float* __restrict__ out) {
  __shared__ float s_c1[NC * 3];
  __shared__ float s_f[NC * 32];                 // 16 KB
  __shared__ alignas(32) _Float16 act[64 * 64];  // feat (59 pad 64)
  __shared__ alignas(32) _Float16 h1b[64 * 32];
  __shared__ alignas(32) _Float16 h2b[64 * 32];
  __shared__ alignas(32) _Float16 h3b[64 * 64];

  int blk = blockIdx.x;
  int b = blk >> 7;                // 128 blocks per batch
  int p0 = (blk & 127) * 64;
  int t = threadIdx.x;

  for (int i = t; i < NC * 3; i += 64) s_c1[i] = c1[(size_t)b * NC * 3 + i];
  for (int i = t; i < NC * 32; i += 64) s_f[i] = c1f[(size_t)b * NC * 32 + i];
  __syncthreads();

  { // phase A: each thread owns one point
    int p = p0 + t;
    const float* xp = x + ((size_t)b * NPTS + p) * 3;
    float xx = xp[0], yy = xp[1], zz = xp[2];
    _Float16* f = act + t * 64;
    f[0] = (_Float16)xx; f[1] = (_Float16)yy; f[2] = (_Float16)zz;
#pragma unroll
    for (int l = 0; l < 4; ++l) {
      float fr = (float)(1 << l);
      f[3 + l * 6 + 0] = (_Float16)sinf(fr * xx);
      f[3 + l * 6 + 1] = (_Float16)sinf(fr * yy);
      f[3 + l * 6 + 2] = (_Float16)sinf(fr * zz);
      f[3 + l * 6 + 3] = (_Float16)cosf(fr * xx);
      f[3 + l * 6 + 4] = (_Float16)cosf(fr * yy);
      f[3 + l * 6 + 5] = (_Float16)cosf(fr * zz);
    }
    float bd[16]; int bi[16];
    for (int k = 0; k < 16; ++k) { bd[k] = 3.4e38f; bi[k] = 0; }
    for (int j = 0; j < NC; ++j) {
      float dx = xx - s_c1[j * 3], dy = yy - s_c1[j * 3 + 1], dz = zz - s_c1[j * 3 + 2];
      float d = dx * dx + dy * dy + dz * dz;
      if (d < bd[15]) {
        int pos = 15;
        while (pos > 0 && bd[pos - 1] > d) {
          bd[pos] = bd[pos - 1]; bi[pos] = bi[pos - 1]; --pos;
        }
        bd[pos] = d; bi[pos] = j;
      }
    }
    float w[16], ws = 0.f;
    for (int k = 0; k < 16; ++k) { w[k] = 1.f / (bd[k] + 1e-4f); ws += w[k]; }
    float inv = 1.f / ws;
    float acc[32];
#pragma unroll
    for (int cch = 0; cch < 32; ++cch) acc[cch] = 0.f;
    for (int k = 0; k < 16; ++k) {
      float ww = w[k] * inv;
      const float* ff = &s_f[bi[k] * 32];
#pragma unroll
      for (int cch = 0; cch < 32; ++cch) acc[cch] += ww * ff[cch];
    }
#pragma unroll
    for (int cch = 0; cch < 32; ++cch) f[27 + cch] = (_Float16)acc[cch];
#pragma unroll
    for (int cch = 59; cch < 64; ++cch) f[cch] = (_Float16)0.f;
  }
  __syncthreads();

  // phase B: wave wv processes tiles 2*wv and 2*wv+1 (16 points each)
  int lane = t & 31;
  int wv = t >> 5;
  for (int tt = 0; tt < 2; ++tt) {
    int row = (wv * 2 + tt) * 16;
    run_layer<64, 32>(act + row * 64, h1b + row * 32, wpack + OFF_P11, biases + BO_P11, lane);
    run_layer<32, 32>(h1b + row * 32, h2b + row * 32, wpack + OFF_P12, biases + BO_P12, lane);
    run_layer<32, 64>(h2b + row * 32, h3b + row * 64, wpack + OFF_H1, biases + BO_H1, lane);

    // head2: 64 -> 16 (3 valid), no relu, straight to d_out
    v8f c = mk_C(biases + BO_H2, 0, lane);
#pragma unroll
    for (int kc = 0; kc < 2; ++kc)
      c = wmma32(ld_A(h3b + row * 64, 64, kc * 32, lane), ld_B(wpack + OFF_H2 + kc * 512, lane), c);
    int col = lane & 15, rbase = (lane & 16) ? 8 : 0;
    if (col < 3) {
#pragma unroll
      for (int r = 0; r < 8; ++r)
        out[((size_t)b * NPTS + p0 + row + rbase + r) * 3 + col] = c[r];
    }
  }
}

// ---------------- host ----------------
extern "C" void kernel_launch(void* const* d_in, const int* in_sizes, int n_in,
                              void* d_out, int out_size, void* d_ws, size_t ws_size,
                              hipStream_t stream) {
  (void)in_sizes; (void)n_in; (void)out_size; (void)ws_size;
  auto F = [&](int i) { return (const float*)d_in[i]; };

  const float* x      = F(0);
  const float* c1     = F(1);
  const float* c1feat = F(2);
  // params pytree (sorted dict keys): abs(3 layers x {W,b,beta,g,m,v}),
  // head {W1,W2,b1,b2}, prop1(2 layers), prop2(2 layers)
  int idx = 3;
  struct L { const float *W, *b, *beta, *g, *m, *v; };
  L absL[3], p1L[2], p2L[2];
  for (int i = 0; i < 3; ++i) { absL[i] = {F(idx), F(idx+1), F(idx+2), F(idx+3), F(idx+4), F(idx+5)}; idx += 6; }
  const float* hW1 = F(idx++); const float* hW2 = F(idx++);
  const float* hb1 = F(idx++); const float* hb2 = F(idx++);
  for (int i = 0; i < 2; ++i) { p1L[i] = {F(idx), F(idx+1), F(idx+2), F(idx+3), F(idx+4), F(idx+5)}; idx += 6; }
  for (int i = 0; i < 2; ++i) { p2L[i] = {F(idx), F(idx+1), F(idx+2), F(idx+3), F(idx+4), F(idx+5)}; idx += 6; }

  char* wsb = (char*)d_ws;
  _Float16* wpack = (_Float16*)(wsb + WS_WPACK);
  float* biases   = (float*)(wsb + WS_BIAS);
  float* c2pos    = (float*)(wsb + WS_C2POS);
  float* c2feat   = (float*)(wsb + WS_C2FEAT);
  float* c1fw     = (float*)(wsb + WS_C1F);

  auto pack = [&](L l, int cin, int cout, int Kp, int Np, int woff, int boff, int bn) {
    int blocks = (Kp * Np + 255) / 256;
    pack_kernel<<<blocks, 256, 0, stream>>>(l.W, l.b, l.beta, l.g, l.m, l.v,
                                            cin, cout, Kp, Np, bn,
                                            wpack + woff, biases + boff);
  };
  pack(absL[0], 384, 64, 384, 64, OFF_ABS1, BO_ABS1, 1);
  pack(absL[1], 64, 32, 64, 32, OFF_ABS2, BO_ABS2, 1);
  pack(absL[2], 32, 12, 32, 16, OFF_ABS3, BO_ABS3, 1);
  pack(p2L[0], 12, 64, 32, 64, OFF_P21, BO_P21, 1);
  pack(p2L[1], 64, 32, 64, 32, OFF_P22, BO_P22, 1);
  pack(p1L[0], 59, 32, 64, 32, OFF_P11, BO_P11, 1);
  pack(p1L[1], 32, 32, 32, 32, OFF_P12, BO_P12, 1);
  pack({hW1, hb1, nullptr, nullptr, nullptr, nullptr}, 32, 64, 32, 64, OFF_H1, BO_H1, 0);
  pack({hW2, hb2, nullptr, nullptr, nullptr, nullptr}, 64, 3, 64, 16, OFF_H2, BO_H2, 0);

  fps_kernel<<<BATCH, NC, 0, stream>>>(c1, c2pos);
  abs_kernel<<<BATCH * NG, 32, 0, stream>>>(c1, c1feat, c2pos, wpack, biases, c2feat);
  prop2_kernel<<<BATCH * (NC / 16), 32, 0, stream>>>(c1, c2pos, c2feat, wpack, biases, c1fw);
  tail_kernel<<<BATCH * NPTS / 64, 64, 0, stream>>>(x, c1, c1fw, wpack, biases, (float*)d_out);
}
